// simple_GAT_regression_14912126452013
// MI455X (gfx1250) — matrix operand order repaired and checked by reading, not compiled, not run on hardware
//
#include <hip/hip_runtime.h>
#include <hip/hip_bf16.h>
#include <math.h>

// ---------------------------------------------------------------------------
// Problem constants (from reference)
// ---------------------------------------------------------------------------
#define NN   20000      // nodes
#define EE   320000     // edges (before self loops)
#define DIN  128        // input channels
#define HH   2          // heads
#define CC   128        // hidden per head
#define DD   256        // H*C
#define GG   64         // graphs
#define NEG_SLOPE 0.2f
#define BN_EPS    1e-5f

typedef __attribute__((ext_vector_type(16))) __bf16 bf16x16;
typedef __attribute__((ext_vector_type(8)))  float  f32x8;

// ---------------------------------------------------------------------------
// WMMA GEMM:  H[n, 256] = X[n, K] @ W[K, 256]   (bf16 inputs, fp32 accum)
// Block: 256 threads = 8 waves; block tile 128(M) x 128(N); k-step 32.
// Wave (4x2 grid) computes 32x64 = 2x4 WMMA 16x16 tiles.
// ---------------------------------------------------------------------------
#define BM 128
#define BN 128
#define BK 32

union FragU { bf16x16 v; uint4 q[2]; };

__global__ __launch_bounds__(256)
void gemm_wmma_kernel(const float* __restrict__ Xp, const float* __restrict__ Wp,
                      float* __restrict__ Hp, int n, int k)
{
    __shared__ alignas(16) __bf16 As[BM][BK];   // [m][k]
    __shared__ alignas(16) __bf16 Bs[BN][BK];   // transposed: [n][k]

    const int tid  = threadIdx.x;
    const int lane = tid & 31;
    const int wave = tid >> 5;
    const int g    = lane >> 4;      // half-wave 0/1
    const int l16  = lane & 15;
    const int wm   = wave >> 1;      // 0..3 along M
    const int wn   = wave & 1;       // 0..1 along N
    const int m0   = blockIdx.x * BM;
    const int n0   = blockIdx.y * BN;

    f32x8 acc[2][4] = {};

    for (int k0 = 0; k0 < k; k0 += BK) {
        // stage A tile (fp32 -> bf16)
        for (int i = tid; i < BM * BK; i += 256) {
            int m  = i >> 5;
            int kk = i & 31;
            int gm = m0 + m;
            float v = (gm < n) ? Xp[(size_t)gm * k + (k0 + kk)] : 0.0f;
            As[m][kk] = (__bf16)v;
        }
        // stage B tile transposed: Bs[nn][kk] = W[k0+kk][n0+nn]
        for (int i = tid; i < BN * BK; i += 256) {
            int nn = i >> 5;
            int kk = i & 31;
            Bs[nn][kk] = (__bf16)Wp[(size_t)(k0 + kk) * DD + (n0 + nn)];
        }
        __syncthreads();

        // A fragments: lane (g,l16): M = base + l16, K = {8g..8g+7} U {16+8g..23+8g}
        FragU afrag[2];
        #pragma unroll
        for (int ti = 0; ti < 2; ++ti) {
            int row = wm * 32 + ti * 16 + l16;
            afrag[ti].q[0] = *reinterpret_cast<const uint4*>(&As[row][8 * g]);
            afrag[ti].q[1] = *reinterpret_cast<const uint4*>(&As[row][16 + 8 * g]);
        }
        // B fragments: lane (g,l16): N = base + l16, K = {8g..8g+7} U {16+8g..23+8g}
        FragU bfrag[4];
        #pragma unroll
        for (int tj = 0; tj < 4; ++tj) {
            int col = wn * 64 + tj * 16 + l16;
            bfrag[tj].q[0] = *reinterpret_cast<const uint4*>(&Bs[col][8 * g]);
            bfrag[tj].q[1] = *reinterpret_cast<const uint4*>(&Bs[col][16 + 8 * g]);
        }

        #pragma unroll
        for (int ti = 0; ti < 2; ++ti)
            #pragma unroll
            for (int tj = 0; tj < 4; ++tj)
                acc[ti][tj] = __builtin_amdgcn_wmma_f32_16x16x32_bf16(
                    false, afrag[ti].v, false, bfrag[tj].v,
                    (short)0, acc[ti][tj], false, false);
        __syncthreads();
    }

    // store: C/D layout — VGPR r -> M = r + 8g, N = l16
    #pragma unroll
    for (int ti = 0; ti < 2; ++ti) {
        #pragma unroll
        for (int tj = 0; tj < 4; ++tj) {
            int col = n0 + wn * 64 + tj * 16 + l16;
            #pragma unroll
            for (int r = 0; r < 8; ++r) {
                int row = m0 + wm * 32 + ti * 16 + r + 8 * g;
                if (row < n) Hp[(size_t)row * DD + col] = acc[ti][tj][r];
            }
        }
    }
}

// ---------------------------------------------------------------------------
// Per-node attention logits: al_s[n,h] = <h[n,h,:], a_src[h,:]>, same for dst
// ---------------------------------------------------------------------------
__global__ void node_alpha_kernel(const float* __restrict__ Hp,
                                  const float* __restrict__ asrc,
                                  const float* __restrict__ adst,
                                  float* __restrict__ al_s, float* __restrict__ al_d)
{
    int idx = blockIdx.x * blockDim.x + threadIdx.x;  // n*H
    if (idx >= NN * HH) return;
    int node = idx >> 1, head = idx & 1;
    const float* hp = Hp + (size_t)node * DD + head * CC;
    const float* a1 = asrc + head * CC;
    const float* a2 = adst + head * CC;
    float s = 0.f, d = 0.f;
    for (int c = 0; c < CC; ++c) { float v = hp[c]; s += v * a1[c]; d += v * a2[c]; }
    al_s[idx] = s; al_d[idx] = d;
}

// ---------------------------------------------------------------------------
// Per-layer init: agg=0, mmax=-inf, denom=0, bn accumulators=0
// ---------------------------------------------------------------------------
__global__ void init_layer_kernel(float* __restrict__ agg, float* __restrict__ mmax,
                                  float* __restrict__ denom, float* __restrict__ bnsum,
                                  float* __restrict__ bnsq)
{
    int idx = blockIdx.x * blockDim.x + threadIdx.x;
    if (idx < NN * DD) agg[idx] = 0.0f;
    if (idx < NN * HH) { mmax[idx] = -3.4e38f; denom[idx] = 0.0f; }
    if (idx < DD) { bnsum[idx] = 0.0f; bnsq[idx] = 0.0f; }
}

__device__ inline void atomicMaxFloat(float* addr, float val)
{
    // IEEE ordering trick; valid with -inf initialization
    if (val >= 0.0f) atomicMax((int*)addr, __float_as_int(val));
    else             atomicMin((unsigned int*)addr, __float_as_uint(val));
}

__device__ inline float leaky(float v) { return v > 0.0f ? v : v * NEG_SLOPE; }

// ---------------------------------------------------------------------------
// Edge pass 1: segment max of leakyrelu(al_s[src]+al_d[dst]) over dst
// ---------------------------------------------------------------------------
__global__ void edge_max_kernel(const int* __restrict__ src, const int* __restrict__ dst,
                                const float* __restrict__ al_s, const float* __restrict__ al_d,
                                float* __restrict__ mmax)
{
    int e = blockIdx.x * blockDim.x + threadIdx.x;
    if (e >= EE + NN) return;
    int s = (e < EE) ? src[e] : (e - EE);
    int d = (e < EE) ? dst[e] : (e - EE);
    atomicMaxFloat(&mmax[d * 2 + 0], leaky(al_s[s * 2 + 0] + al_d[d * 2 + 0]));
    atomicMaxFloat(&mmax[d * 2 + 1], leaky(al_s[s * 2 + 1] + al_d[d * 2 + 1]));
}

// ---------------------------------------------------------------------------
// Edge pass 2: ex = exp(e - m[dst]); denom[dst] += ex (store ex for pass 3)
// ---------------------------------------------------------------------------
__global__ void edge_exp_kernel(const int* __restrict__ src, const int* __restrict__ dst,
                                const float* __restrict__ al_s, const float* __restrict__ al_d,
                                const float* __restrict__ mmax,
                                float* __restrict__ exbuf, float* __restrict__ denom)
{
    int e = blockIdx.x * blockDim.x + threadIdx.x;
    if (e >= EE + NN) return;
    int s = (e < EE) ? src[e] : (e - EE);
    int d = (e < EE) ? dst[e] : (e - EE);
    #pragma unroll
    for (int h = 0; h < HH; ++h) {
        float v  = leaky(al_s[s * 2 + h] + al_d[d * 2 + h]);
        float ex = __expf(v - mmax[d * 2 + h]);
        exbuf[(size_t)e * 2 + h] = ex;
        atomicAdd(&denom[d * 2 + h], ex);
    }
}

// ---------------------------------------------------------------------------
// Edge pass 3: agg[dst,:] += (ex/denom[dst])[head(c)] * h[src,c].  1 wave/edge.
// ---------------------------------------------------------------------------
__global__ __launch_bounds__(256)
void edge_agg_kernel(const int* __restrict__ src, const int* __restrict__ dst,
                     const float* __restrict__ exbuf, const float* __restrict__ denom,
                     const float* __restrict__ Hp, float* __restrict__ agg)
{
    int e = blockIdx.x * 8 + (threadIdx.x >> 5);
    if (e >= EE + NN) return;
    int lane = threadIdx.x & 31;
    int s = (e < EE) ? src[e] : (e - EE);
    int d = (e < EE) ? dst[e] : (e - EE);
    float a0 = exbuf[(size_t)e * 2 + 0] / (denom[d * 2 + 0] + 1e-16f);
    float a1 = exbuf[(size_t)e * 2 + 1] / (denom[d * 2 + 1] + 1e-16f);
    const float* hs = Hp  + (size_t)s * DD;
    float*       od = agg + (size_t)d * DD;
    __builtin_prefetch(hs + lane, 0, 0);   // global_prefetch_b8
    #pragma unroll
    for (int j = 0; j < 8; ++j) {
        int c = lane + j * 32;
        float a = (c < CC) ? a0 : a1;
        atomicAdd(&od[c], a * hs[c]);
    }
}

// ---------------------------------------------------------------------------
// relu(agg + bias) -> xbuf ; per-channel sum / sumsq (one block per channel)
// ---------------------------------------------------------------------------
__global__ __launch_bounds__(256)
void bn_stats_kernel(const float* __restrict__ agg, const float* __restrict__ bias,
                     float* __restrict__ xbuf, float* __restrict__ bnsum,
                     float* __restrict__ bnsq)
{
    int d = blockIdx.x;          // 0..255
    int t = threadIdx.x;
    float b = bias[d];
    float s = 0.f, s2 = 0.f;
    for (int row = t; row < NN; row += 256) {
        float y = agg[(size_t)row * DD + d] + b;
        y = y > 0.0f ? y : 0.0f;
        xbuf[(size_t)row * DD + d] = y;
        s += y; s2 += y * y;
    }
    __shared__ float sh[256], sh2[256];
    sh[t] = s; sh2[t] = s2;
    __syncthreads();
    for (int st = 128; st > 0; st >>= 1) {
        if (t < st) { sh[t] += sh[t + st]; sh2[t] += sh2[t + st]; }
        __syncthreads();
    }
    if (t == 0) { bnsum[d] = sh[0]; bnsq[d] = sh2[0]; }
}

__global__ void bn_apply_kernel(float* __restrict__ xbuf, const float* __restrict__ bnsum,
                                const float* __restrict__ bnsq, const float* __restrict__ gamma,
                                const float* __restrict__ beta)
{
    int idx = blockIdx.x * blockDim.x + threadIdx.x;
    if (idx >= NN * DD) return;
    int d = idx & (DD - 1);
    float mu  = bnsum[d] * (1.0f / NN);
    float var = bnsq[d] * (1.0f / NN) - mu * mu;
    xbuf[idx] = (xbuf[idx] - mu) * rsqrtf(var + BN_EPS) * gamma[d] + beta[d];
}

// ---------------------------------------------------------------------------
// Pooling + regression head
// ---------------------------------------------------------------------------
__global__ void pool_init_kernel(float* __restrict__ featsum, float* __restrict__ cnt)
{
    int idx = blockIdx.x * blockDim.x + threadIdx.x;
    if (idx < GG * DD) featsum[idx] = 0.0f;
    if (idx < GG) cnt[idx] = 0.0f;
}

__global__ void pool_kernel(const float* __restrict__ xbuf, const int* __restrict__ batch,
                            float* __restrict__ featsum, float* __restrict__ cnt)
{
    int idx = blockIdx.x * blockDim.x + threadIdx.x;
    if (idx >= NN * DD) return;
    int n = idx >> 8, d = idx & (DD - 1);
    int b = batch[n];
    atomicAdd(&featsum[b * DD + d], xbuf[idx]);
    if (d == 0) atomicAdd(&cnt[b], 1.0f);
}

__global__ __launch_bounds__(256)
void head_kernel(const float* __restrict__ featsum, const float* __restrict__ cnt,
                 const float* __restrict__ regW, const float* __restrict__ regb,
                 float* __restrict__ out)
{
    int g = blockIdx.x;          // 0..63
    int t = threadIdx.x;         // 0..255 == D
    float c = cnt[g]; c = c < 1.0f ? 1.0f : c;
    float f = featsum[g * DD + t] / c;
    out[GG + g * DD + t] = f;                    // feat (second output)
    __shared__ float sh[256];
    sh[t] = f * regW[t];
    __syncthreads();
    for (int st = 128; st > 0; st >>= 1) {
        if (t < st) sh[t] += sh[t + st];
        __syncthreads();
    }
    if (t == 0) out[g] = sh[0] + regb[0];        // pred (first output)
}

// ---------------------------------------------------------------------------
// Host launcher
// ---------------------------------------------------------------------------
extern "C" void kernel_launch(void* const* d_in, const int* in_sizes, int n_in,
                              void* d_out, int out_size, void* d_ws, size_t ws_size,
                              hipStream_t stream)
{
    const float* X      = (const float*)d_in[0];
    const int*   ei     = (const int*)  d_in[1];
    const int*   batch  = (const int*)  d_in[2];
    const float* W1     = (const float*)d_in[3];
    const float* as1    = (const float*)d_in[4];
    const float* ad1    = (const float*)d_in[5];
    const float* b1     = (const float*)d_in[6];
    const float* W25    = (const float*)d_in[7];
    const float* as25   = (const float*)d_in[8];
    const float* ad25   = (const float*)d_in[9];
    const float* b25    = (const float*)d_in[10];
    const float* gammas = (const float*)d_in[11];
    const float* betas  = (const float*)d_in[12];
    const float* regW   = (const float*)d_in[13];
    const float* regb   = (const float*)d_in[14];
    float* out = (float*)d_out;

    const int Etot = EE + NN;

    // workspace layout (floats)
    float* ws = (float*)d_ws;
    size_t o = 0;
    float* hbuf    = ws + o; o += (size_t)NN * DD;
    float* agg     = ws + o; o += (size_t)NN * DD;
    float* xbuf    = ws + o; o += (size_t)NN * DD;
    float* al_s    = ws + o; o += (size_t)NN * HH;
    float* al_d    = ws + o; o += (size_t)NN * HH;
    float* mmax    = ws + o; o += (size_t)NN * HH;
    float* denom   = ws + o; o += (size_t)NN * HH;
    float* exbuf   = ws + o; o += (size_t)Etot * HH;
    float* bnsum   = ws + o; o += DD;
    float* bnsq    = ws + o; o += DD;
    float* featsum = ws + o; o += (size_t)GG * DD;
    float* cnt     = ws + o; o += GG;

    const int* srcArr = ei;
    const int* dstArr = ei + EE;

    for (int l = 0; l < 5; ++l) {
        const float* xin = (l == 0) ? X   : xbuf;
        const int    K   = (l == 0) ? DIN : DD;
        const float* W   = (l == 0) ? W1  : W25 + (size_t)(l - 1) * DD * DD;
        const float* av  = (l == 0) ? as1 : as25 + (size_t)(l - 1) * DD;
        const float* adv = (l == 0) ? ad1 : ad25 + (size_t)(l - 1) * DD;
        const float* bv  = (l == 0) ? b1  : b25  + (size_t)(l - 1) * DD;

        dim3 ggrid((NN + BM - 1) / BM, DD / BN);
        gemm_wmma_kernel<<<ggrid, 256, 0, stream>>>(xin, W, hbuf, NN, K);
        node_alpha_kernel<<<(NN * HH + 255) / 256, 256, 0, stream>>>(hbuf, av, adv, al_s, al_d);
        init_layer_kernel<<<(NN * DD + 255) / 256, 256, 0, stream>>>(agg, mmax, denom, bnsum, bnsq);
        edge_max_kernel<<<(Etot + 255) / 256, 256, 0, stream>>>(srcArr, dstArr, al_s, al_d, mmax);
        edge_exp_kernel<<<(Etot + 255) / 256, 256, 0, stream>>>(srcArr, dstArr, al_s, al_d, mmax, exbuf, denom);
        edge_agg_kernel<<<(Etot + 7) / 8, 256, 0, stream>>>(srcArr, dstArr, exbuf, denom, hbuf, agg);
        bn_stats_kernel<<<DD, 256, 0, stream>>>(agg, bv, xbuf, bnsum, bnsq);
        bn_apply_kernel<<<(NN * DD + 255) / 256, 256, 0, stream>>>(xbuf, bnsum, bnsq,
                                                                   gammas + l * DD, betas + l * DD);
    }

    pool_init_kernel<<<(GG * DD + 255) / 256, 256, 0, stream>>>(featsum, cnt);
    pool_kernel<<<(NN * DD + 255) / 256, 256, 0, stream>>>(xbuf, batch, featsum, cnt);
    head_kernel<<<GG, 256, 0, stream>>>(featsum, cnt, regW, regb, out);
}
